// MambaTower_36490042147416
// MI455X (gfx1250) — matrix operand order repaired and checked by reading, not compiled, not run on hardware
//
#include <hip/hip_runtime.h>
#include <hip/hip_bf16.h>

// ---------------- constants ----------------
#define L_   2048
#define DM_  256
#define E_   512
#define N_   16
#define R_   16
#define K_   4
#define T_   8
#define DT_  64
#define H_   4
#define DH_  64

typedef __attribute__((ext_vector_type(16))) _Float16 v16h;
typedef __attribute__((ext_vector_type(8)))  _Float16 v8h;
typedef __attribute__((ext_vector_type(8)))  float    v8f;

// ---------------- small helpers ----------------
__device__ __forceinline__ float silu_f(float x) {
  return x / (1.0f + __expf(-x));
}

// ---------------- f32 -> f16 conversion (row-major, activations) ----------------
__global__ void cvt_f32_f16(const float* __restrict__ src, _Float16* __restrict__ dst, int n) {
  int i = blockIdx.x * blockDim.x + threadIdx.x;
  if (i < n) dst[i] = (_Float16)src[i];
}

// f32 (K x N row-major) -> f16 transposed (N x K): weight staging for WMMA B side.
// Lane's B fragment (fixed col, k = 8h+0..7 and 8h+16..23) becomes two aligned b128 loads.
__global__ void cvt_f32_f16_T(const float* __restrict__ src, _Float16* __restrict__ dst,
                              int K, int N) {
  int i = blockIdx.x * blockDim.x + threadIdx.x;   // over N*K, k fastest in dst
  if (i >= K * N) return;
  int k   = i % K;
  int col = i / K;
  dst[i] = (_Float16)src[(size_t)k * N + col];
}

// tracks (T, L, DT) -> A matrix rows (l*T + t), cols d, in f16
__global__ void cvt_tracks_f16(const float* __restrict__ src, _Float16* __restrict__ dst) {
  int i = blockIdx.x * blockDim.x + threadIdx.x;      // L*T*DT = 1M
  if (i >= L_ * T_ * DT_) return;
  int d = i & (DT_ - 1);
  int t = (i >> 6) & (T_ - 1);
  int l = i >> 9;
  dst[i] = (_Float16)src[((size_t)t * L_ + l) * DT_ + d];
}

// ---------------- RMSNorm (fused f16 emit) ----------------
__global__ void rmsnorm_f16(const float* __restrict__ x, const float* __restrict__ w,
                            _Float16* __restrict__ out) {
  __shared__ float red[8];
  int row = blockIdx.x;
  int c   = threadIdx.x;                  // 256 threads, one per DM column
  float v = x[(size_t)row * DM_ + c];
  float s = v * v;
  #pragma unroll
  for (int off = 16; off > 0; off >>= 1) s += __shfl_xor(s, off, 32);
  if ((c & 31) == 0) red[c >> 5] = s;
  __syncthreads();
  float tot = 0.f;
  #pragma unroll
  for (int i = 0; i < 8; ++i) tot += red[i];
  float scale = rsqrtf(tot / (float)DM_ + 1e-6f);
  out[(size_t)row * DM_ + c] = (_Float16)(v * scale * w[c]);
}

// ---------------- WMMA GEMM, 16 x (16*NT) strip per wave ----------------
// C[M x N] = A[M x K] (f16 row-major) * B (f16, staged TRANSPOSED: N x K) (+bias)(+res)
// NT is a compile-time tile count so all NT accumulators live in fixed VGPRs
// (no movrel): per 32-deep k-step -> one A fragment, NT back-to-back v_wmma.
template <int NT>
__global__ void gemm_wmma_t(const _Float16* __restrict__ A, const _Float16* __restrict__ Bt,
                            const float* __restrict__ bias, const float* __restrict__ res,
                            float* __restrict__ C, int M, int N, int Kd, int n_base) {
  int lane = threadIdx.x;                 // 32 lanes
  int m0 = blockIdx.y * 16;
  int n0 = n_base + blockIdx.x * (16 * NT);
  int half = lane >> 4;                   // 0 | 1
  int l16  = lane & 15;
  int arow = m0 + l16;
  v8f acc[NT];
  #pragma unroll
  for (int tt = 0; tt < NT; ++tt) acc[tt] = (v8f){};
  for (int k0 = 0; k0 < Kd; k0 += 32) {
    // ---- A fragment (16-bit A 16x32 layout): two aligned b128 loads ----
    const _Float16* ap = A + (size_t)arow * Kd + k0 + half * 8;
    __builtin_prefetch(ap + 32, 0, 1);    // global_prefetch_b8 next k-slab
    v8h a1 = *(const v8h*)(ap);           // k = +0..7   (rel. to half offset)
    v8h a2 = *(const v8h*)(ap + 16);      // k = +16..23
    v16h a;
    #pragma unroll
    for (int j = 0; j < 8; ++j) { a[j] = a1[j]; a[8 + j] = a2[j]; }
    // ---- NT B fragments from transposed B: contiguous per lane ----
    #pragma unroll
    for (int tt = 0; tt < NT; ++tt) {
      int col = n0 + tt * 16 + l16;
      const _Float16* bp = Bt + (size_t)col * Kd + k0 + half * 8;
      v8h b1 = *(const v8h*)(bp);         // k = +0..7   (rel. to half offset)
      v8h b2 = *(const v8h*)(bp + 16);    // k = +16..23
      v16h b;
      #pragma unroll
      for (int j = 0; j < 8; ++j) { b[j] = b1[j]; b[8 + j] = b2[j]; }
      acc[tt] = __builtin_amdgcn_wmma_f32_16x16x32_f16(false, a, false, b,
                                                       (short)0, acc[tt], false, false);
    }
  }
  // ---- epilogue: C/D 16x16 f32 layout ----
  #pragma unroll
  for (int tt = 0; tt < NT; ++tt) {
    int bcol = n0 + tt * 16 + l16;
    #pragma unroll
    for (int r = 0; r < 8; ++r) {
      int row = m0 + half * 8 + r;
      float v = acc[tt][r];
      if (bias) v += bias[bcol];
      size_t idx = (size_t)row * N + bcol;
      if (res)  v += res[idx];
      C[idx] = v;
    }
  }
}

// ---------------- depthwise causal conv (both directions) + SiLU ----------------
// u2[dir*L + l, e]; dir1 operates on time-flipped xin (stored in flipped order).
__global__ void conv_silu(const float* __restrict__ xz, const float* __restrict__ cw,
                          const float* __restrict__ cb, float* __restrict__ u2) {
  int i = blockIdx.x * blockDim.x + threadIdx.x;   // 2*L*E = 2^21
  if (i >= 2 * L_ * E_) return;
  int e   = i & (E_ - 1);
  int l   = (i >> 9) & (L_ - 1);
  int dir = i >> 20;
  float acc = cb[e];
  #pragma unroll
  for (int k = 0; k < K_; ++k) {
    int src = (dir == 0) ? (l - (K_ - 1) + k) : (L_ - 1 - l + (K_ - 1) - k);
    if (src >= 0 && src < L_)
      acc += xz[(size_t)src * (2 * E_) + e] * cw[k * E_ + e];
  }
  u2[i] = silu_f(acc);
}

// ---------------- dt = softplus(pr[:, :R] @ dw + db) ----------------
__global__ void dt_softplus(const float* __restrict__ pr2, const float* __restrict__ dw,
                            const float* __restrict__ db, float* __restrict__ dt2) {
  int i = blockIdx.x * blockDim.x + threadIdx.x;   // 2*L*E
  if (i >= 2 * L_ * E_) return;
  int e   = i & (E_ - 1);
  int row = i >> 9;
  float acc = db[e];
  #pragma unroll
  for (int r = 0; r < R_; ++r)
    acc += pr2[(size_t)row * 48 + r] * dw[r * E_ + e];
  dt2[i] = (acc > 20.f) ? acc : log1pf(__expf(acc));
}

// ---------------- selective scan (fwd + bwd), lane = (e, n) ----------------
__global__ void ssm_scan(const float* __restrict__ u2, const float* __restrict__ dt2,
                         const float* __restrict__ pr2, const float* __restrict__ alog,
                         const float* __restrict__ Dp, float* __restrict__ y) {
  int gid = (blockIdx.x * blockDim.x + threadIdx.x) >> 4;  // channel e (512 groups)
  int n   = threadIdx.x & 15;
  if (gid >= E_) return;
  int e = gid;
  float A = -__expf(alog[e * N_ + n]);
  float D = Dp[e];
  for (int dir = 0; dir < 2; ++dir) {
    float s = 0.f;
    for (int l = 0; l < L_; ++l) {
      int row   = dir * L_ + l;
      float dtv = dt2[(size_t)row * E_ + e];
      float u   = u2[(size_t)row * E_ + e];
      float Bv  = pr2[(size_t)row * 48 + R_ + n];
      float Cv  = pr2[(size_t)row * 48 + R_ + N_ + n];
      s = s * __expf(dtv * A) + dtv * Bv * u;
      float c = s * Cv;
      c += __shfl_xor(c, 1, 32);
      c += __shfl_xor(c, 2, 32);
      c += __shfl_xor(c, 4, 32);
      c += __shfl_xor(c, 8, 32);
      if (n == 0) {
        int ol = (dir == 0) ? l : (L_ - 1 - l);
        float val = c + u * D;
        if (dir == 0) y[(size_t)ol * E_ + e]  = val;
        else          y[(size_t)ol * E_ + e] += val;
      }
    }
  }
}

// ---------------- y * silu(z) -> f16 (A staging for out_proj) ----------------
__global__ void ygate_f16(const float* __restrict__ y, const float* __restrict__ xz,
                          _Float16* __restrict__ a16) {
  int i = blockIdx.x * blockDim.x + threadIdx.x;   // L*E
  if (i >= L_ * E_) return;
  int e = i & (E_ - 1);
  int l = i >> 9;
  float z = xz[(size_t)l * (2 * E_) + E_ + e];
  a16[i] = (_Float16)(y[i] * silu_f(z));
}

// ---------------- cross-attention over T=8 tracks per position ----------------
__global__ void cross_attn(const float* __restrict__ Q, const float* __restrict__ Kb,
                           const float* __restrict__ Vb, float* __restrict__ out) {
  __shared__ float sQ[DM_];
  __shared__ float sK[T_ * DM_];
  __shared__ float sV[T_ * DM_];
  __shared__ float slog[H_ * T_];
  int l = blockIdx.x;
  int c = threadIdx.x;                               // 256
  sQ[c] = Q[(size_t)l * DM_ + c];
  #pragma unroll
  for (int t = 0; t < T_; ++t) {
    sK[t * DM_ + c] = Kb[((size_t)l * T_ + t) * DM_ + c];
    sV[t * DM_ + c] = Vb[((size_t)l * T_ + t) * DM_ + c];
  }
  __syncthreads();
  int h = c >> 6;
  int d = c & 63;
  if (d < T_) {
    int t = d;
    float acc = 0.f;
    #pragma unroll
    for (int j = 0; j < DH_; ++j)
      acc += sQ[h * DH_ + j] * sK[t * DM_ + h * DH_ + j];
    slog[h * T_ + t] = acc * 0.125f;                 // 1/sqrt(64)
  }
  __syncthreads();
  float mx = -1e30f;
  #pragma unroll
  for (int t = 0; t < T_; ++t) mx = fmaxf(mx, slog[h * T_ + t]);
  float w[T_], wsum = 0.f;
  #pragma unroll
  for (int t = 0; t < T_; ++t) { w[t] = __expf(slog[h * T_ + t] - mx); wsum += w[t]; }
  float inv = 1.f / wsum;
  float acc = 0.f;
  #pragma unroll
  for (int t = 0; t < T_; ++t) acc += w[t] * sV[t * DM_ + c];
  out[(size_t)l * DM_ + c] = acc * inv;
}

// ---------------- host orchestration ----------------
extern "C" void kernel_launch(void* const* d_in, const int* in_sizes, int n_in,
                              void* d_out, int out_size, void* d_ws, size_t ws_size,
                              hipStream_t stream) {
  const float* x_in    = (const float*)d_in[0];
  const float* tracks  = (const float*)d_in[1];
  const float* mnorm   = (const float*)d_in[2];
  const float* inproj  = (const float*)d_in[3];
  const float* convw   = (const float*)d_in[4];
  const float* convb   = (const float*)d_in[5];
  const float* xprojw  = (const float*)d_in[6];
  const float* dtw     = (const float*)d_in[7];
  const float* dtb     = (const float*)d_in[8];
  const float* alog    = (const float*)d_in[9];
  const float* Dp      = (const float*)d_in[10];
  const float* outpw   = (const float*)d_in[11];
  const float* canorm  = (const float*)d_in[12];
  const float* qw      = (const float*)d_in[13];
  const float* qb      = (const float*)d_in[14];
  const float* kw      = (const float*)d_in[15];
  const float* kb      = (const float*)d_in[16];
  const float* vw      = (const float*)d_in[17];
  const float* vb      = (const float*)d_in[18];
  const float* ow      = (const float*)d_in[19];
  const float* ob      = (const float*)d_in[20];

  float* xcur = (float*)d_out;                       // running residual stream

  char* ws = (char*)d_ws;
  _Float16* w16  = (_Float16*)(ws + 0x0000000);      // 512 KB  weight staging (transposed)
  _Float16* a16  = (_Float16*)(ws + 0x0100000);      // 4 MB    activation staging
  float*    xz   = (float*)   (ws + 0x0500000);      // 8 MB    L x 1024
  float*    u2   = (float*)   (ws + 0x0D00000);      // 8 MB    2L x 512
  float*    dt2  = (float*)   (ws + 0x1500000);      // 8 MB    2L x 512
  float*    pr2  = (float*)   (ws + 0x1D00000);      // 1 MB    2L x 48
  float*    ybuf = (float*)   (ws + 0x1E00000);      // 4 MB    L x 512
  // attention scratch overlaps dead mamba scratch
  float*    Qb   = (float*)   (ws + 0x0500000);      // 2 MB
  float*    att  = (float*)   (ws + 0x0700000);      // 2 MB
  float*    Kbuf = (float*)   (ws + 0x0D00000);      // 16 MB (u2+dt2 region)
  float*    Vbuf = (float*)   (ws + 0x1D00000);      // 16 MB (pr2+ybuf+...)

  hipMemcpyAsync(xcur, x_in, (size_t)L_ * DM_ * sizeof(float),
                 hipMemcpyDeviceToDevice, stream);

  auto cvt = [&](const float* s, _Float16* d, int n) {
    cvt_f32_f16<<<(n + 255) / 256, 256, 0, stream>>>(s, d, n);
  };
  auto cvtT = [&](const float* s, _Float16* d, int K, int N) {
    int n = K * N;
    cvt_f32_f16_T<<<(n + 255) / 256, 256, 0, stream>>>(s, d, K, N);
  };
  // Dispatch: full 64-wide column blocks as NT=4; remainder column as NT=1..3.
  auto gemm = [&](const _Float16* A, const _Float16* Bt, const float* bias,
                  const float* res, float* C, int M, int N, int K) {
    int fullCols = N / 64;
    int rem = (N % 64) / 16;
    if (fullCols > 0)
      gemm_wmma_t<4><<<dim3(fullCols, M / 16), 32, 0, stream>>>(
          A, Bt, bias, res, C, M, N, K, 0);
    if (rem == 1)
      gemm_wmma_t<1><<<dim3(1, M / 16), 32, 0, stream>>>(
          A, Bt, bias, res, C, M, N, K, fullCols * 64);
    else if (rem == 2)
      gemm_wmma_t<2><<<dim3(1, M / 16), 32, 0, stream>>>(
          A, Bt, bias, res, C, M, N, K, fullCols * 64);
    else if (rem == 3)
      gemm_wmma_t<3><<<dim3(1, M / 16), 32, 0, stream>>>(
          A, Bt, bias, res, C, M, N, K, fullCols * 64);
  };

  for (int i = 0; i < 4; ++i) {
    // ---- Mamba layer i ----
    rmsnorm_f16<<<L_, DM_, 0, stream>>>(xcur, mnorm + i * DM_, a16);
    cvtT(inproj + (size_t)i * DM_ * 2 * E_, w16, DM_, 2 * E_);
    gemm(a16, w16, nullptr, nullptr, xz, L_, 2 * E_, DM_);                 // xz = h @ wi

    conv_silu<<<(2 * L_ * E_) / 256, 256, 0, stream>>>(
        xz, convw + (size_t)i * K_ * E_, convb + (size_t)i * E_, u2);

    cvt(u2, a16, 2 * L_ * E_);
    cvtT(xprojw + (size_t)i * E_ * 48, w16, E_, 48);
    gemm(a16, w16, nullptr, nullptr, pr2, 2 * L_, 48, E_);                 // pr = u @ xw

    dt_softplus<<<(2 * L_ * E_) / 256, 256, 0, stream>>>(
        pr2, dtw + (size_t)i * R_ * E_, dtb + (size_t)i * E_, dt2);

    ssm_scan<<<(E_ * 16) / 256, 256, 0, stream>>>(
        u2, dt2, pr2, alog + (size_t)i * E_ * N_, Dp + (size_t)i * E_, ybuf);

    ygate_f16<<<(L_ * E_) / 256, 256, 0, stream>>>(ybuf, xz, a16);
    cvtT(outpw + (size_t)i * E_ * DM_, w16, E_, DM_);
    gemm(a16, w16, nullptr, xcur, xcur, L_, DM_, E_);                      // x += y' @ wo

    // ---- cross-attention after layers 1 and 3 ----
    if (i == 1 || i == 3) {
      int j = (i == 1) ? 0 : 1;
      rmsnorm_f16<<<L_, DM_, 0, stream>>>(xcur, canorm + j * DM_, a16);
      cvtT(qw + (size_t)j * DM_ * DM_, w16, DM_, DM_);
      gemm(a16, w16, qb + j * DM_, nullptr, Qb, L_, DM_, DM_);             // Q

      cvt_tracks_f16<<<(L_ * T_ * DT_) / 256, 256, 0, stream>>>(tracks, a16);
      cvtT(kw + (size_t)j * DT_ * DM_, w16, DT_, DM_);
      gemm(a16, w16, kb + j * DM_, nullptr, Kbuf, L_ * T_, DM_, DT_);      // K
      cvtT(vw + (size_t)j * DT_ * DM_, w16, DT_, DM_);
      gemm(a16, w16, vb + j * DM_, nullptr, Vbuf, L_ * T_, DM_, DT_);      // V

      cross_attn<<<L_, DM_, 0, stream>>>(Qb, Kbuf, Vbuf, att);

      cvt(att, a16, L_ * DM_);
      cvtT(ow + (size_t)j * DM_ * DM_, w16, DM_, DM_);
      gemm(a16, w16, ob + j * DM_, xcur, xcur, L_, DM_, DM_);              // x += out @ ow + ob
    }
  }
}
